// STTran_70308614635934
// MI455X (gfx1250) — compile-verified, matrix operand order checked by reading, not run on hardware
//
#include <hip/hip_runtime.h>
#include <hip/hip_bf16.h>

typedef __attribute__((ext_vector_type(16))) __bf16 v16bf;
typedef __attribute__((ext_vector_type(8)))  float  v8f;
typedef int v4i_ __attribute__((vector_size(16)));   // GCC-style int4 (builtin param type)
#define GAS __attribute__((address_space(1)))
#define LAS __attribute__((address_space(3)))

#define N_OBJ 65536
#define PER 64
#define KPAD 384          // 192 feat + 35 dist + 128 pos + 29 pad
#define HID 1024
#define NCPAD 48          // 36 logits padded to 48 (3 x 16 col tiles)
#define ROWS 32           // rows per workgroup in the fused MLP

// Dynamic LDS layout (bytes)
#define SM_HS   0                         // 32*1024 bf16 = 65536
#define SM_AS   65536                     // 32*384  bf16 = 24576
#define SM_LS   (65536 + 24576)           // 32*48   f32  = 6144
#define SM_TOT  (65536 + 24576 + 6144)    // 96256 B (< 320 KB WGP LDS)

__device__ __forceinline__ unsigned short f2bf(float x) {
    union { float f; unsigned u; } c; c.f = x;
    unsigned u = c.u;
    unsigned r = (u + 0x7fffu + ((u >> 16) & 1u)) >> 16;   // round-to-nearest-even
    return (unsigned short)r;
}

// ---------------------------------------------------------------------------
// Kernel 1: fused + BN-folded decoder weight, transposed to N-major bf16.
//   k in [0,192)   : features rows of dec1_W
//   k in [192,227) : (obj_embed_W @ dec1_W[192:392]) -- embedding collapsed
//   k in [227,355) : pos rows of dec1_W (392..519)
//   k in [355,384) : zero pad
// ---------------------------------------------------------------------------
__global__ __launch_bounds__(KPAD) void prep_w1(
    const float* __restrict__ dec1_W, const float* __restrict__ dec1_b,
    const float* __restrict__ bg, const float* __restrict__ bb,
    const float* __restrict__ bm, const float* __restrict__ bv,
    const float* __restrict__ objW,
    unsigned short* __restrict__ WbT, float* __restrict__ b1f)
{
    int j = blockIdx.x, k = threadIdx.x;
    __shared__ float s_s;
    if (k == 0) {
        float s = bg[j] * rsqrtf(bv[j] + 1e-5f);
        s_s = s;
        b1f[j] = (dec1_b[j] - bm[j]) * s + bb[j];
    }
    __syncthreads();
    float s = s_s, val;
    if (k < 192) {
        val = dec1_W[(size_t)k * HID + j];
    } else if (k < 227) {
        int i = k - 192;
        float acc = 0.f;
        for (int e = 0; e < 200; e++)
            acc += objW[i * 200 + e] * dec1_W[(size_t)(192 + e) * HID + j];
        val = acc;
    } else if (k < 355) {
        val = dec1_W[(size_t)(392 + (k - 227)) * HID + j];
    } else {
        val = 0.f;
    }
    WbT[(size_t)j * KPAD + k] = f2bf(val * s);
}

// ---------------------------------------------------------------------------
// Kernel 2: classifier weight transposed, padded to 48 cols, bf16.
// ---------------------------------------------------------------------------
__global__ __launch_bounds__(256) void prep_w2(
    const float* __restrict__ dec2_W, const float* __restrict__ dec2_b,
    unsigned short* __restrict__ W2T, float* __restrict__ b2f)
{
    int j = blockIdx.x;
    for (int k = threadIdx.x; k < HID; k += 256)
        W2T[(size_t)j * HID + k] = f2bf(j < 36 ? dec2_W[(size_t)k * 36 + j] : 0.f);
    if (threadIdx.x == 0) b2f[j] = (j < 36) ? dec2_b[j] : 0.f;
}

// ---------------------------------------------------------------------------
// Kernel 3: per-object bf16 activation row [features | distribution | pos | 0]
// ---------------------------------------------------------------------------
__global__ __launch_bounds__(128) void build_act(
    const float* __restrict__ distn, const float* __restrict__ feat,
    const float* __restrict__ boxes,
    const float* __restrict__ posW, const float* __restrict__ posb,
    const float* __restrict__ g4, const float* __restrict__ b4,
    const float* __restrict__ m4, const float* __restrict__ v4,
    unsigned short* __restrict__ Act)
{
    int o = blockIdx.x, t = threadIdx.x;
    float x1 = boxes[(size_t)o * 5 + 1], y1 = boxes[(size_t)o * 5 + 2];
    float x2 = boxes[(size_t)o * 5 + 3], y2 = boxes[(size_t)o * 5 + 4];
    float w = x2 - x1 + 1.f, h = y2 - y1 + 1.f;
    float cs[4] = { x1 + 0.5f * w, y1 + 0.5f * h, w, h };
    float hb[4];
    #pragma unroll
    for (int c = 0; c < 4; c++)
        hb[c] = (cs[c] - m4[c]) * rsqrtf(v4[c] + 1e-5f) * g4[c] + b4[c];
    float p = hb[0] * posW[t] + hb[1] * posW[128 + t] +
              hb[2] * posW[256 + t] + hb[3] * posW[384 + t] + posb[t];
    p = fmaxf(p, 0.f);
    unsigned short* row = Act + (size_t)o * KPAD;
    row[227 + t] = f2bf(p);
    for (int k = t; k < 192; k += 128) row[k] = f2bf(feat[(size_t)o * 192 + k]);
    if (t < 35) row[192 + t] = f2bf(distn[(size_t)o * 35 + t]);
    if (t < 29) row[355 + t] = 0;
}

// ---------------------------------------------------------------------------
// Kernel 4: fused MLP, 32 rows / block, 8 waves, dynamic LDS (96 KB).
//  Phase 2: hidden = relu(Act @ WbT + b1f)  (bf16 WMMA, K=384)
//           each wave: 2 row-tiles x 8 col-tiles; B fragment double-buffered
//           and reused across both row tiles (halves L2 weight traffic).
//  Phase 3: logits = hidden @ W2T + b2f     (bf16 WMMA, K=1024, LDS-resident)
//  Phase 4: softmax + per-row argmax.
// ---------------------------------------------------------------------------
union Frag { uint4 q[2]; v16bf v; };

__device__ __forceinline__ void loadB(Frag& f, const unsigned short* base,
                                      int k0, int half) {
    f.q[0] = *(const uint4*)(base + k0 + half * 8);
    f.q[1] = *(const uint4*)(base + k0 + 16 + half * 8);
}

__global__ __launch_bounds__(256) void mlp_fused(
    const unsigned short* __restrict__ Act, const unsigned short* __restrict__ WbT,
    const float* __restrict__ b1f,
    const unsigned short* __restrict__ W2T, const float* __restrict__ b2f,
    float* __restrict__ out_dist, float* __restrict__ out_score,
    float* __restrict__ out_label)
{
    extern __shared__ char smem[];
    unsigned short* Hs = (unsigned short*)(smem + SM_HS);   // 32 x 1024 bf16
    unsigned short* As = (unsigned short*)(smem + SM_AS);   // 32 x 384  bf16
    float*          Ls = (float*)(smem + SM_LS);            // 32 x 48   f32

    int tid = threadIdx.x;
    int rt  = blockIdx.x;

    // ---- Stage 32-row activation tile into LDS (async path if available) ----
    {
        const unsigned short* gsrc = Act + (size_t)rt * ROWS * KPAD;
#if defined(__gfx1250__) && __has_builtin(__builtin_amdgcn_global_load_async_to_lds_b128)
        #pragma unroll
        for (int it = 0; it < (ROWS * KPAD * 2 / 16) / 256; it++) {
            int idx = tid + it * 256;                       // one b128 per lane
            __builtin_amdgcn_global_load_async_to_lds_b128(
                (GAS v4i_*)(gsrc + idx * 8),
                (LAS v4i_*)(As + idx * 8),
                0, 0);
        }
    #if __has_builtin(__builtin_amdgcn_s_wait_asynccnt)
        __builtin_amdgcn_s_wait_asynccnt(0);
    #else
        asm volatile("s_wait_asynccnt 0" ::: "memory");
    #endif
#else
        const uint4* src = (const uint4*)gsrc;
        uint4* dst = (uint4*)As;
        #pragma unroll
        for (int it = 0; it < (ROWS * KPAD * 2 / 16) / 256; it++)
            dst[tid + it * 256] = src[tid + it * 256];
#endif
    }
    __syncthreads();

    int w = tid >> 5, lane = tid & 31;
    int m = lane & 15, half = lane >> 4;   // A layout: lanes 0-15 K{0..7,16..23}, 16-31 K{8..15,24..31}
    int jbase = w * 128;                   // each wave owns 128 hidden columns

    v8f acc[16] = {};                      // acc[2*ct + r], r = row tile
    const unsigned short* A0 = As + m * KPAD;          // rows  0..15
    const unsigned short* A1 = As + (16 + m) * KPAD;   // rows 16..31

    for (int ks = 0; ks < 12; ks++) {
        int k0 = ks * 32;
        Frag a0, a1;
        loadB(a0, A0, k0, half);           // ds_load_b128 x2
        loadB(a1, A1, k0, half);
        Frag bf[2];
        loadB(bf[0], WbT + (size_t)(jbase + m) * KPAD, k0, half);
        #pragma unroll
        for (int ct = 0; ct < 8; ct++) {
            if (ct < 7)                    // prefetch next B tile into other buffer
                loadB(bf[(ct + 1) & 1],
                      WbT + (size_t)(jbase + (ct + 1) * 16 + m) * KPAD, k0, half);
            acc[2 * ct + 0] = __builtin_amdgcn_wmma_f32_16x16x32_bf16(
                false, a0.v, false, bf[ct & 1].v, (short)0, acc[2 * ct + 0], false, false);
            acc[2 * ct + 1] = __builtin_amdgcn_wmma_f32_16x16x32_bf16(
                false, a1.v, false, bf[ct & 1].v, (short)0, acc[2 * ct + 1], false, false);
        }
    }

    // Epilogue: bias (BN folded) + relu -> bf16 hidden tile in LDS.
    #pragma unroll
    for (int ct = 0; ct < 8; ct++) {
        int col = jbase + ct * 16 + m;
        float bb = b1f[col];
        #pragma unroll
        for (int r = 0; r < 2; r++) {
            #pragma unroll
            for (int vi = 0; vi < 8; vi++) {
                int row = r * 16 + vi + 8 * half;   // C layout: M = vgpr + 8*(lane>=16)
                float val = fmaxf(acc[2 * ct + r][vi] + bb, 0.f);
                Hs[row * HID + col] = f2bf(val);
            }
        }
    }
    __syncthreads();

    // Phase 3: classifier GEMM. 6 waves: 2 row-tiles x 3 col-tiles, K = 1024.
    if (w < 6) {
        int rbase = (w & 1) * 16;
        int nb    = (w >> 1) * 16;
        const unsigned short* Hrow = Hs + (rbase + m) * HID;
        const unsigned short* Bcol = W2T + (size_t)(nb + m) * HID;
        v8f acc2 = {};
        for (int ks = 0; ks < 32; ks++) {
            int k0 = ks * 32;
            Frag a, b;
            loadB(a, Hrow, k0, half);
            loadB(b, Bcol, k0, half);
            acc2 = __builtin_amdgcn_wmma_f32_16x16x32_bf16(
                false, a.v, false, b.v, (short)0, acc2, false, false);
        }
        int col = nb + m;
        float bb = b2f[col];
        #pragma unroll
        for (int vi = 0; vi < 8; vi++)
            Ls[(rbase + vi + 8 * half) * NCPAD + col] = acc2[vi] + bb;
    }
    __syncthreads();

    // Phase 4: softmax over logits[1:36], outputs.
    if (tid < ROWS) {
        int gr = rt * ROWS + tid;
        float L[36];
        #pragma unroll
        for (int j = 0; j < 36; j++) L[j] = Ls[tid * NCPAD + j];
        float mx = -1e30f;
        #pragma unroll
        for (int j = 1; j < 36; j++) mx = fmaxf(mx, L[j]);
        float d[35], sum = 0.f;
        #pragma unroll
        for (int j = 1; j < 36; j++) { d[j - 1] = __expf(L[j] - mx); sum += d[j - 1]; }
        float inv = 1.f / sum;
        float* od = out_dist + (size_t)gr * 35;
        float best = -1.f; int bi = 1;
        #pragma unroll
        for (int i = 0; i < 35; i++) {
            float di = d[i] * inv;
            od[i] = di;
            if (i >= 1 && di > best) { best = di; bi = i; }
        }
        out_score[gr] = best;
        out_label[gr] = (float)(bi + 1);   // argmax over dist[:,1:] + 2
    }
}

// ---------------------------------------------------------------------------
// Kernel 5: per-image (64 objects) argmax of dist[:,0] -> human overrides.
// ---------------------------------------------------------------------------
__global__ __launch_bounds__(64) void human_fix(
    const float* __restrict__ out_dist, float* __restrict__ out_score,
    float* __restrict__ out_label, float* __restrict__ out_hidx)
{
    __shared__ float sv[64];
    __shared__ int   si[64];
    int b = blockIdx.x, t = threadIdx.x;
    int row = b * PER + t;
    sv[t] = out_dist[(size_t)row * 35 + 0];
    si[t] = t;
    __syncthreads();
    for (int off = 32; off > 0; off >>= 1) {
        if (t < off) {
            float v2 = sv[t + off]; int i2 = si[t + off];
            if (v2 > sv[t] || (v2 == sv[t] && i2 < si[t])) { sv[t] = v2; si[t] = i2; }
        }
        __syncthreads();
    }
    if (t == 0) {
        int hrow = b * PER + si[0];
        out_hidx[b] = (float)hrow;
        out_label[hrow] = 1.0f;
        out_score[hrow] = out_dist[(size_t)hrow * 35 + 0];
    }
}

// ---------------------------------------------------------------------------
extern "C" void kernel_launch(void* const* d_in, const int* in_sizes, int n_in,
                              void* d_out, int out_size, void* d_ws, size_t ws_size,
                              hipStream_t stream)
{
    const float* distribution = (const float*)d_in[0];
    const float* features     = (const float*)d_in[1];
    const float* boxes        = (const float*)d_in[2];
    const float* obj_embed_W  = (const float*)d_in[3];
    const float* bn4_g        = (const float*)d_in[4];
    const float* bn4_b        = (const float*)d_in[5];
    const float* bn4_m        = (const float*)d_in[6];
    const float* bn4_v        = (const float*)d_in[7];
    const float* pos_W        = (const float*)d_in[8];
    const float* pos_b        = (const float*)d_in[9];
    const float* dec1_W       = (const float*)d_in[10];
    const float* dec1_b       = (const float*)d_in[11];
    const float* bn1k_g       = (const float*)d_in[12];
    const float* bn1k_b       = (const float*)d_in[13];
    const float* bn1k_m       = (const float*)d_in[14];
    const float* bn1k_v       = (const float*)d_in[15];
    const float* dec2_W       = (const float*)d_in[16];
    const float* dec2_b       = (const float*)d_in[17];

    // Workspace layout
    char* ws = (char*)d_ws;
    unsigned short* Act = (unsigned short*)ws;                         // N*384 bf16
    unsigned short* WbT = (unsigned short*)(ws + (size_t)N_OBJ * KPAD * 2);
    float*          b1f = (float*)((char*)WbT + (size_t)HID * KPAD * 2);
    unsigned short* W2T = (unsigned short*)((char*)b1f + HID * 4);
    float*          b2f = (float*)((char*)W2T + (size_t)NCPAD * HID * 2);

    float* out      = (float*)d_out;
    float* o_dist   = out;                                  // N*35
    float* o_score  = out + (size_t)N_OBJ * 35;             // N
    float* o_label  = out + (size_t)N_OBJ * 36;             // N
    float* o_hidx   = out + (size_t)N_OBJ * 37;             // N/PER

    prep_w1<<<HID, KPAD, 0, stream>>>(dec1_W, dec1_b, bn1k_g, bn1k_b, bn1k_m, bn1k_v,
                                      obj_embed_W, WbT, b1f);
    prep_w2<<<NCPAD, 256, 0, stream>>>(dec2_W, dec2_b, W2T, b2f);
    build_act<<<N_OBJ, 128, 0, stream>>>(distribution, features, boxes,
                                         pos_W, pos_b, bn4_g, bn4_b, bn4_m, bn4_v, Act);
    mlp_fused<<<N_OBJ / ROWS, 256, SM_TOT, stream>>>(Act, WbT, b1f, W2T, b2f,
                                                     o_dist, o_score, o_label);
    human_fix<<<N_OBJ / PER, PER, 0, stream>>>(o_dist, o_score, o_label, o_hidx);
}
